// GGNN_22557168239479
// MI455X (gfx1250) — matrix operand order repaired and checked by reading, not compiled, not run on hardware
//
#include <hip/hip_runtime.h>

typedef __attribute__((ext_vector_type(16))) _Float16 v16h;
typedef __attribute__((ext_vector_type(8)))  float    v8f;

#define D   256
#define KT  8      // K tiles: 256 / 32

// ---------------------------------------------------------------- utilities
__global__ void k_zero_f32(float* __restrict__ p, long n) {
    long i = (long)blockIdx.x * blockDim.x + threadIdx.x;
    if (i < n) p[i] = 0.f;
}

// h[n, :] = embed_w[node_ids[n], :]   (float4 vectorized)
__global__ void k_embed(const int* __restrict__ ids, const float* __restrict__ emb,
                        float* __restrict__ h, int N) {
    long i = (long)blockIdx.x * blockDim.x + threadIdx.x;   // over N * 64 float4
    if (i >= (long)N * (D / 4)) return;
    int n  = (int)(i >> 6);
    int d4 = (int)(i & 63);
    const float4* src = (const float4*)(emb + (size_t)ids[n] * D);
    ((float4*)(h + (size_t)n * D))[d4] = src[d4];
}

// per-edge-type scalar weight = mean over D of edge_embed_w[t, :]
__global__ void k_type_mean(const float* __restrict__ ee, float* __restrict__ tmean) {
    __shared__ float red[256];
    int t = blockIdx.x;                      // 0..6
    red[threadIdx.x] = ee[(size_t)t * D + threadIdx.x];
    __syncthreads();
    for (int s = 128; s > 0; s >>= 1) {
        if ((int)threadIdx.x < s) red[threadIdx.x] += red[threadIdx.x + s];
        __syncthreads();
    }
    if (threadIdx.x == 0) tmean[t] = red[0] * (1.f / 256.f);
}

// ------------------------------------------------- operand fragment packing
// A fragment (16x32 f16, ISA 7.12.2): lane holds row M = lane%16,
// element j -> K = (j<8?0:16) + (lane>>4)*8 + (j&7).  One thread per
// (mTile,kTile,lane): 4 coalesced float4 loads -> one 32B v16h store.
__global__ void k_pack_a(const float* __restrict__ A, _Float16* __restrict__ Ap,
                         int N, int mTiles) {
    long gid = (long)blockIdx.x * blockDim.x + threadIdx.x;
    long total = (long)mTiles * KT * 32;
    if (gid >= total) return;
    int lane = (int)(gid & 31);
    long rem = gid >> 5;
    int kt = (int)(rem % KT);
    int mt = (int)(rem / KT);
    int row  = mt * 16 + (lane & 15);
    int half = lane >> 4;
    v16h frag;
    if (row < N) {
        const float* ar = A + (size_t)row * D + kt * 32 + half * 8;
        #pragma unroll
        for (int j = 0; j < 8; ++j) frag[j]     = (_Float16)ar[j];
        #pragma unroll
        for (int j = 0; j < 8; ++j) frag[8 + j] = (_Float16)ar[16 + j];
    } else {
        #pragma unroll
        for (int j = 0; j < 16; ++j) frag[j] = (_Float16)0.f;
    }
    *(v16h*)(Ap + ((size_t)(mt * KT + kt) * 32 + lane) * 16) = frag;
}

// B fragment (32x16 f16): lane holds col N = lane%16,
// element j -> K = (j<8?0:16) + (lane>>4)*8 + (j&7).
// transpose==0: W is [K, ncols] row-major (ggnn layer weight)
// transpose==1: W is [n, 256] row-major, B[k][n] = W[n*256 + k] (gru w^T)
__global__ void k_pack_b(const float* __restrict__ W, _Float16* __restrict__ Bp,
                         int kTiles, int nTiles, int transpose) {
    long gid = (long)blockIdx.x * blockDim.x + threadIdx.x;
    long total = (long)kTiles * nTiles * 32;
    if (gid >= total) return;
    int lane = (int)(gid & 31);
    long rem = gid >> 5;
    int nt = (int)(rem % nTiles);
    int kt = (int)(rem / nTiles);
    int n     = nt * 16 + (lane & 15);
    int kbase = kt * 32 + 8 * (lane >> 4);
    int ncols = nTiles * 16;
    v16h frag;
    #pragma unroll
    for (int j = 0; j < 16; ++j) {
        int k = kbase + (j < 8 ? j : j + 8);
        float v = transpose ? W[(size_t)n * D + k] : W[(size_t)k * ncols + n];
        frag[j] = (_Float16)v;
    }
    *(v16h*)(Bp + ((size_t)(kt * nTiles + nt) * 32 + lane) * 16) = frag;
}

// ------------------------------------------------------------- WMMA GEMM
// C[M, ...] (+)= A[M,256] * B[:, nTileOff*16 .. ] (+ bias).  One wave
// computes a 16x64 output tile: A fragment reused across 4 WMMAs.
// NTB (B-panel nTile count) is a template constant so every fragment load
// folds to an immediate offset off one base pointer (24-bit IOFFSET).
template <int NTB>
__global__ void k_gemm_wmma(const _Float16* __restrict__ Ap,
                            const _Float16* __restrict__ Bp,
                            float* __restrict__ C,
                            int mTiles, int N,
                            int nTileOff, int nTileCnt,
                            int ldc, const float* __restrict__ bias,
                            int accumulate) {
    int wave = (int)((blockIdx.x * blockDim.x + threadIdx.x) >> 5);
    int lane = threadIdx.x & 31;
    int nGroups = nTileCnt >> 2;
    if (wave >= mTiles * nGroups) return;
    int mt = wave / nGroups;
    int ng = wave % nGroups;

    v8f acc[4] = {};
    const v16h* Abase = (const v16h*)Ap + (size_t)mt * KT * 32 + lane;
    const v16h* Bbase = (const v16h*)Bp + (size_t)(nTileOff + ng * 4) * 32 + lane;

    #pragma unroll
    for (int kt = 0; kt < KT; ++kt) {
        v16h a = Abase[kt * 32];                       // imm offset kt*1024B
        #pragma unroll
        for (int t = 0; t < 4; ++t) {
            v16h b = Bbase[(kt * NTB + t) * 32];       // imm offset, const fold
            acc[t] = __builtin_amdgcn_wmma_f32_16x16x32_f16(
                false, a, false, b, (short)0, acc[t], false, false);
        }
    }

    int rowBase = mt * 16 + ((lane >> 4) << 3);   // r in VGPR -> row = base + r
    int colLane = lane & 15;
    #pragma unroll
    for (int t = 0; t < 4; ++t) {
        int nloc = ng * 4 + t;
        int col  = nloc * 16 + colLane;
        float bv = bias ? bias[(size_t)(nTileOff + nloc) * 16 + colLane] : 0.f;
        #pragma unroll
        for (int r = 0; r < 8; ++r) {
            int row = rowBase + r;
            if (row < N) {
                size_t idx = (size_t)row * ldc + col;
                float v = acc[t][r] + bv;
                if (accumulate) C[idx] += v; else C[idx] = v;
            }
        }
    }
}

// ------------------------------------------------ edge gather/scale/scatter
// one wave per edge: agg[dst, :] += m[src, :] * tmean[type-1].
// src/dst/w are wave-uniform -> readfirstlane puts them in SGPRs so the
// gather and the atomics use SGPR-base addressing.
__global__ void k_scatter(const float* __restrict__ m, const int* __restrict__ edges,
                          const int* __restrict__ et, const float* __restrict__ tmean,
                          float* __restrict__ agg, int E) {
    long gid = (long)blockIdx.x * blockDim.x + threadIdx.x;
    int e    = (int)(gid >> 5);
    int lane = (int)(gid & 31);
    if (e >= E) return;
    int src = __builtin_amdgcn_readfirstlane(edges[e]);
    int dst = __builtin_amdgcn_readfirstlane(edges[(size_t)E + e]);
    int wi  = __builtin_amdgcn_readfirstlane(__float_as_int(tmean[et[e] - 1]));
    float w = __int_as_float(wi);
    const float4* ms = (const float4*)(m + (size_t)src * D);
    float* ad = agg + (size_t)dst * D;
    #pragma unroll
    for (int i = 0; i < 2; ++i) {
        int d4 = lane + 32 * i;
        float4 v = ms[d4];
        atomicAdd(ad + d4 * 4 + 0, v.x * w);
        atomicAdd(ad + d4 * 4 + 1, v.y * w);
        atomicAdd(ad + d4 * 4 + 2, v.z * w);
        atomicAdd(ad + d4 * 4 + 3, v.w * w);
    }
}

// -------------------------------------------------------- fused GRU gates
// G[n,0:256]   = i_r + h_r (+biases), G[n,256:512] = i_z + h_z,
// G[n,512:768] = i_n (+b_ih), agg[n,:] = h_n (+b_hh).  Re-zeros agg.
__global__ void k_gru(const float* __restrict__ G, float* __restrict__ agg,
                      float* __restrict__ h, int N) {
    long i = (long)blockIdx.x * blockDim.x + threadIdx.x;
    if (i >= (long)N * D) return;
    int n = (int)(i >> 8), d = (int)(i & 255);
    size_t gb = (size_t)n * 768;
    float rs = G[gb + d];
    float zs = G[gb + 256 + d];
    float in = G[gb + 512 + d];
    float hn = agg[i];
    float r  = 1.f / (1.f + __expf(-rs));
    float z  = 1.f / (1.f + __expf(-zs));
    float nn = tanhf(in + r * hn);
    h[i]   = (1.f - z) * nn + z * h[i];
    agg[i] = 0.f;                       // ready for next layer's scatter
}

// ------------------------------------------------------ attention pooling
// one wave per node: s[n] = exp(sigmoid(x[n].gate_w + gate_b)); total += s[n]
__global__ void k_gate(const float* __restrict__ h, const float* __restrict__ gw,
                       const float* __restrict__ gb, float* __restrict__ s,
                       float* __restrict__ total, int N) {
    long gid = (long)blockIdx.x * blockDim.x + threadIdx.x;
    int n = (int)(gid >> 5), lane = (int)(gid & 31);
    if (n >= N) return;
    const float4* x = (const float4*)(h + (size_t)n * D);
    const float4* g = (const float4*)gw;
    float acc = 0.f;
    #pragma unroll
    for (int i = 0; i < 2; ++i) {
        float4 a = x[lane + 32 * i], b = g[lane + 32 * i];
        acc += a.x * b.x + a.y * b.y + a.z * b.z + a.w * b.w;
    }
    for (int off = 16; off; off >>= 1) acc += __shfl_xor(acc, off, 32);
    if (lane == 0) {
        float gate = 1.f / (1.f + __expf(-(acc + gb[0])));
        float e = __expf(gate);
        s[n] = e;
        atomicAdd(total, e);
    }
}

// hg[d] = sum_n s[n] * h[n, d]   (unnormalized; divided in k_out)
__global__ void k_wsum(const float* __restrict__ h, const float* __restrict__ s,
                       float* __restrict__ hg, int N) {
    int d = threadIdx.x;                         // 256 threads
    float acc = 0.f;
    for (int n = blockIdx.x; n < N; n += gridDim.x)
        acc += s[n] * h[(size_t)n * D + d];
    atomicAdd(&hg[d], acc);
}

__global__ void k_out(const float* __restrict__ hg, const float* __restrict__ total,
                      const float* __restrict__ ow, const float* __restrict__ ob,
                      float* __restrict__ out, int C) {
    int c = blockIdx.x * blockDim.x + threadIdx.x;
    if (c >= C) return;
    float inv = 1.f / total[0];
    float acc = 0.f;
    for (int k = 0; k < D; ++k) acc += hg[k] * ow[(size_t)c * D + k];
    out[c] = acc * inv + ob[c];
}

// ---------------------------------------------------------------- launcher
static inline unsigned cdiv(long a, long b) { return (unsigned)((a + b - 1) / b); }

extern "C" void kernel_launch(void* const* d_in, const int* in_sizes, int n_in,
                              void* d_out, int out_size, void* d_ws, size_t ws_size,
                              hipStream_t stream) {
    const int*   node_ids = (const int*)  d_in[0];
    const int*   edges    = (const int*)  d_in[1];
    const int*   etypes   = (const int*)  d_in[2];
    const float* embed_w  = (const float*)d_in[3];
    const float* edge_ew  = (const float*)d_in[4];
    const float* ggnn_w   = (const float*)d_in[5];
    const float* w_ih     = (const float*)d_in[6];
    const float* w_hh     = (const float*)d_in[7];
    const float* b_ih     = (const float*)d_in[8];
    const float* b_hh     = (const float*)d_in[9];
    const float* gate_w   = (const float*)d_in[10];
    const float* gate_b   = (const float*)d_in[11];
    const float* out_w    = (const float*)d_in[12];
    const float* out_b    = (const float*)d_in[13];
    float* out = (float*)d_out;

    const int N = in_sizes[0];
    const int E = in_sizes[2];
    const int L = in_sizes[5] / (D * D);
    const int mTiles = (N + 15) / 16;
    const int nClasses = out_size;

    // ---- workspace carve-up (256B aligned slices) ----
    char* p = (char*)d_ws;
    auto carve = [&](size_t bytes) { char* r = p; p += (bytes + 255) & ~(size_t)255; return r; };
    float*    h     = (float*)   carve((size_t)N * D * 4);       // hidden state
    float*    agg   = (float*)   carve((size_t)N * D * 4);       // msg agg / h_n
    float*    G     = (float*)   carve((size_t)N * 3 * D * 4);   // m (first N*D) then gi
    float*    sbuf  = (float*)   carve((size_t)N * 4);           // exp(gate)
    float*    hg    = (float*)   carve(257 * 4);                 // hg[256] + total
    float*    total = hg + 256;
    float*    tmean = (float*)   carve(32);                      // 7 floats
    _Float16* h_p   = (_Float16*)carve((size_t)mTiles * KT * 512 * 2);
    _Float16* a_p   = (_Float16*)carve((size_t)mTiles * KT * 512 * 2);
    _Float16* wg_p  = (_Float16*)carve((size_t)L * KT * 16 * 512 * 2);
    _Float16* wih_p = (_Float16*)carve((size_t)KT * 48 * 512 * 2);
    _Float16* whh_p = (_Float16*)carve((size_t)KT * 48 * 512 * 2);
    (void)ws_size; (void)n_in;

    // ---- setup ----
    k_zero_f32<<<cdiv((long)N * D, 256), 256, 0, stream>>>(agg, (long)N * D);
    k_embed   <<<cdiv((long)N * (D / 4), 256), 256, 0, stream>>>(node_ids, embed_w, h, N);
    k_type_mean<<<7, 256, 0, stream>>>(edge_ew, tmean);
    for (int l = 0; l < L; ++l)
        k_pack_b<<<cdiv((long)KT * 16 * 32, 256), 256, 0, stream>>>(
            ggnn_w + (size_t)l * D * D, wg_p + (size_t)l * KT * 16 * 512, KT, 16, 0);
    k_pack_b<<<cdiv((long)KT * 48 * 32, 256), 256, 0, stream>>>(w_ih, wih_p, KT, 48, 1);
    k_pack_b<<<cdiv((long)KT * 48 * 32, 256), 256, 0, stream>>>(w_hh, whh_p, KT, 48, 1);

    // ---- GGNN layers ----
    for (int l = 0; l < L; ++l) {
        k_pack_a<<<cdiv((long)mTiles * KT * 32, 256), 256, 0, stream>>>(h, h_p, N, mTiles);
        // m = h @ W_l                (into first N*D of G, ldc=256)
        k_gemm_wmma<16><<<cdiv((long)mTiles * 4 * 32, 256), 256, 0, stream>>>(
            h_p, wg_p + (size_t)l * KT * 16 * 512, G,
            mTiles, N, 0, 16, D, nullptr, 0);
        // agg[dst] += m[src] * edge_weight
        k_scatter<<<cdiv((long)E * 32, 256), 256, 0, stream>>>(G, edges, etypes, tmean, agg, E);
        k_pack_a<<<cdiv((long)mTiles * KT * 32, 256), 256, 0, stream>>>(agg, a_p, N, mTiles);
        // gi = agg @ w_ih^T + b_ih  (all 48 n-tiles, ldc=768)
        k_gemm_wmma<48><<<cdiv((long)mTiles * 12 * 32, 256), 256, 0, stream>>>(
            a_p, wih_p, G, mTiles, N, 0, 48, 3 * D, b_ih, 0);
        // gi[:, 0:512] += h @ w_hh^T[:, 0:512] + b_hh[0:512]   (r,z sums)
        k_gemm_wmma<48><<<cdiv((long)mTiles * 8 * 32, 256), 256, 0, stream>>>(
            h_p, whh_p, G, mTiles, N, 0, 32, 3 * D, b_hh, 1);
        // agg = h @ w_hh^T[:, 512:768] + b_hh[512:768]          (h_n part)
        k_gemm_wmma<48><<<cdiv((long)mTiles * 4 * 32, 256), 256, 0, stream>>>(
            h_p, whh_p, agg, mTiles, N, 32, 16, D, b_hh, 0);
        // GRU gates; re-zeros agg for next layer
        k_gru<<<cdiv((long)N * D, 256), 256, 0, stream>>>(G, agg, h, N);
    }

    // ---- global attention pooling + classifier ----
    k_zero_f32<<<2, 256, 0, stream>>>(hg, 257);
    k_gate <<<cdiv((long)N * 32, 256), 256, 0, stream>>>(h, gate_w, gate_b, sbuf, total, N);
    k_wsum <<<200, 256, 0, stream>>>(h, sbuf, hg, N);
    k_out  <<<1, 128, 0, stream>>>(hg, total, out_w, out_b, out, nClasses);
}
